// MultiSubwordAttentionTagger_86663850099017
// MI455X (gfx1250) — matrix operand order repaired and checked by reading, not compiled, not run on hardware
//
#include <hip/hip_runtime.h>

#define N_SEQ 1024
#define BATCH 16
#define EDIM  256
#define EMB   768
#define HID   512
#define TAG   32

#if defined(__has_builtin)
#  if __has_builtin(__builtin_amdgcn_global_load_async_to_lds_b128)
#    define HAVE_ASYNC_LDS 1
#  endif
#endif
#ifndef HAVE_ASYNC_LDS
#  define HAVE_ASYNC_LDS 0
#endif

typedef __attribute__((ext_vector_type(16))) __bf16 v16bf;
typedef __attribute__((ext_vector_type(8)))  float  v8f;

#if HAVE_ASYNC_LDS
typedef int v4i_vs __attribute__((vector_size(16)));
typedef __attribute__((address_space(1))) v4i_vs* gptr_b128;   // global src
typedef __attribute__((address_space(3))) v4i_vs* lptr_b128;   // LDS dst
#endif

struct U8 { unsigned int u[8]; };

__device__ __forceinline__ unsigned short f2bf(float f) {
  unsigned int u = __builtin_bit_cast(unsigned int, f);
  u += 0x7FFFu + ((u >> 16) & 1u);            // round-to-nearest-even
  return (unsigned short)(u >> 16);
}
__device__ __forceinline__ float bf2f(unsigned short s) {
  unsigned int u = ((unsigned int)s) << 16;
  return __builtin_bit_cast(float, u);
}

// A-matrix (16x32 bf16) K index for VGPR v, per ISA 7.12.2
__device__ __forceinline__ int a_kmap(int v, int lane) {
  int kb = (lane < 16) ? 0 : 8;
  return (v < 4) ? (kb + 2 * v) : (kb + 16 + 2 * (v - 4));
}

// A (16x32) fragment from row-major bf16 (row stride ld elements)
__device__ __forceinline__ v16bf load_a_bf16(const unsigned short* src, int ld) {
  int lane = threadIdx.x & 31;
  const unsigned short* row = src + (size_t)(lane & 15) * ld;
  U8 r;
#pragma unroll
  for (int v = 0; v < 8; ++v) {
    int k = a_kmap(v, lane);
    r.u[v] = (unsigned)row[k] | ((unsigned)row[k + 1] << 16);
  }
  return __builtin_bit_cast(v16bf, r);
}

// B (32x16) fragment, source N-major: element (K,N) at src[n*ld + K]
__device__ __forceinline__ v16bf load_b_nk(const unsigned short* src, int ld) {
  int lane = threadIdx.x & 31;
  const unsigned short* col = src + (size_t)(lane & 15) * ld;
  U8 r;
#pragma unroll
  for (int v = 0; v < 8; ++v) {
    int k = a_kmap(v, lane);
    r.u[v] = (unsigned)col[k] | ((unsigned)col[k + 1] << 16);
  }
  return __builtin_bit_cast(v16bf, r);
}

__device__ __forceinline__ v8f wmma_bf16(v16bf a, v16bf b, v8f c) {
  return __builtin_amdgcn_wmma_f32_16x16x32_bf16(false, a, false, b, (short)0, c,
                                                 false, false);
}

// ---------------------------------------------------------------------------
// P0: weights -> bf16 (transposed N-major), zero Frobenius accumulators
__global__ void k_prep_w(const float* __restrict__ W1, const float* __restrict__ W2,
                         unsigned short* __restrict__ W1t,
                         unsigned short* __restrict__ W2t, float* __restrict__ sumsq) {
  int i = blockIdx.x * blockDim.x + threadIdx.x;
  if (i < EMB * HID) {                 // W_x2h (768x512) -> W1t (512x768)
    int k = i / HID, o = i % HID;
    W1t[(size_t)o * EMB + k] = f2bf(W1[i]);
  } else {
    int j = i - EMB * HID;             // W_hg2z (1536x32) -> W2t (32x1536)
    if (j < 3 * HID * TAG) {
      int k = j / TAG, o = j % TAG;
      W2t[(size_t)o * (3 * HID) + k] = f2bf(W2[j]);
    }
  }
  if (i < 4) sumsq[i] = 0.0f;
}

// ---------------------------------------------------------------------------
// P1: gather embeddings -> bf16 concat rows; build L2-normalized xn0/xn1 (B,N,E)
__global__ void k_gather(const int* __restrict__ xc, const int* __restrict__ x0,
                         const int* __restrict__ x1, const float* __restrict__ Ec,
                         const float* __restrict__ E0, const float* __restrict__ E1,
                         unsigned short* __restrict__ embed,
                         unsigned short* __restrict__ xn0,
                         unsigned short* __restrict__ xn1) {
  int m = blockIdx.x;          // m = n*B + b
  int n = m >> 4, b = m & 15;
  int t = threadIdx.x;         // 0..255
  float c  = Ec[(size_t)xc[m] * EDIM + t];
  float s0 = E0[(size_t)x0[m] * EDIM + t];
  float s1 = E1[(size_t)x1[m] * EDIM + t];
  size_t eb = (size_t)m * EMB;
  embed[eb + t]            = f2bf(c);
  embed[eb + EDIM + t]     = f2bf(s0);
  embed[eb + 2 * EDIM + t] = f2bf(s1);

  __shared__ float red[2][8];
  float q0 = s0 * s0, q1 = s1 * s1;
#pragma unroll
  for (int o = 16; o > 0; o >>= 1) {
    q0 += __shfl_xor(q0, o, 32);
    q1 += __shfl_xor(q1, o, 32);
  }
  if ((t & 31) == 0) { red[0][t >> 5] = q0; red[1][t >> 5] = q1; }
  __syncthreads();
  float n0 = 0.0f, n1 = 0.0f;
#pragma unroll
  for (int i = 0; i < 8; ++i) { n0 += red[0][i]; n1 += red[1][i]; }
  float inv0 = rsqrtf(n0), inv1 = rsqrtf(n1);
  size_t xb = ((size_t)b * N_SEQ + n) * EDIM + t;
  xn0[xb] = f2bf(s0 * inv0);
  xn1[xb] = f2bf(s1 * inv1);
}

// ---------------------------------------------------------------------------
// K1: h = embed @ W_x2h + b (WMMA bf16). Writes h twice:
//   h_att (B, N, HID)  - A-operand layout for the final GEMM
//   ht    (B, HID, N)  - N-major B-operand layout for P@h in attention
__global__ void k_hgemm(const unsigned short* __restrict__ embed,
                        const unsigned short* __restrict__ W1t,
                        const float* __restrict__ bias,
                        unsigned short* __restrict__ h_att,
                        unsigned short* __restrict__ ht) {
  int n = blockIdx.x;                  // m-tile == sequence position (rows = batches)
  int wave = threadIdx.x >> 5, lane = threadIdx.x & 31;
  const unsigned short* arow = embed + (size_t)n * 16 * EMB;

  v8f acc[8];
#pragma unroll
  for (int oi = 0; oi < 8; ++oi)
    for (int r = 0; r < 8; ++r) acc[oi][r] = 0.0f;

  for (int kt = 0; kt < EMB / 32; ++kt) {
    v16bf a = load_a_bf16(arow + kt * 32, EMB);       // one A per K-step
    __builtin_prefetch(arow + (kt + 2) * 32, 0, 3);
    v16bf bf[8];                                      // deep load clause first,
#pragma unroll
    for (int oi = 0; oi < 8; ++oi)
      bf[oi] = load_b_nk(W1t + (size_t)(wave + oi * 4) * 16 * EMB + kt * 32, EMB);
#pragma unroll
    for (int oi = 0; oi < 8; ++oi)                    // then the WMMA chain
      acc[oi] = wmma_bf16(a, bf[oi], acc[oi]);
  }
#pragma unroll
  for (int oi = 0; oi < 8; ++oi) {
    int col = (wave + oi * 4) * 16 + (lane & 15);
    float bv = bias[col];
#pragma unroll
    for (int r = 0; r < 8; ++r) {
      int bidx = r + ((lane < 16) ? 0 : 8);           // M index == batch
      unsigned short hv = f2bf(acc[oi][r] + bv);
      h_att[((size_t)bidx * N_SEQ + n) * HID + col] = hv;
      ht[((size_t)bidx * HID + col) * N_SEQ + n]    = hv;
    }
  }
}

// ---------------------------------------------------------------------------
// K2: one (16-row block, batch, attention): scores -> softmax -> P@h
// Dynamic LDS: S (16x1024 f32, 64KB) + Pbf (16x1024 bf16, 32KB) = 96KB/WG
__global__ void k_attn(const unsigned short* __restrict__ xnA,
                       const unsigned short* __restrict__ xnB,
                       const unsigned short* __restrict__ ht,
                       unsigned short* __restrict__ g0,
                       unsigned short* __restrict__ g1,
                       float* __restrict__ sumsq) {
  extern __shared__ __align__(16) char smem_raw[];
  float*          S   = (float*)smem_raw;                              // 64 KB
  unsigned short* Pbf = (unsigned short*)(smem_raw + 16 * N_SEQ * 4);  // 32 KB

  int it = blockIdx.x;     // 0..63 row blocks
  int b  = blockIdx.y;     // 0..15 batch
  int z  = blockIdx.z;     // 0..1  which attention
  const unsigned short* xn = (z == 0 ? xnA : xnB) + (size_t)b * N_SEQ * EDIM;
  unsigned short* g = (z == 0 ? g0 : g1) + ((size_t)b * N_SEQ + it * 16) * HID;
  int wave = threadIdx.x >> 5, lane = threadIdx.x & 31;

  // Phase A: S[16][1024] = xn_block @ xn^T (K=256); A-fragments hoisted per wave
  {
    const unsigned short* arow = xn + (size_t)it * 16 * EDIM;
    const unsigned short* asrc = arow;
#if HAVE_ASYNC_LDS
    {
      // Stage the 16x256 bf16 query block (8 KB) into LDS once per workgroup
      // via gfx1250 async DMA (ASYNCcnt), reusing the Pbf area temporarily.
      const char* gsrc = (const char*)arow;
      char* ldst = (char*)Pbf;
      int t = threadIdx.x;
#pragma unroll
      for (int i = 0; i < 4; ++i) {
        int off = (i * 128 + t) * 16;                 // 128 thr x 16 B x 4
        __builtin_amdgcn_global_load_async_to_lds_b128(
            (gptr_b128)(gsrc + off), (lptr_b128)(ldst + off), 0, 0);
      }
      asm volatile("s_wait_asynccnt 0" ::: "memory");
      __syncthreads();
      asrc = (const unsigned short*)Pbf;              // ds_load fragments
    }
#endif
    v16bf af[8];
#pragma unroll
    for (int kt = 0; kt < EDIM / 32; ++kt) af[kt] = load_a_bf16(asrc + kt * 32, EDIM);

    for (int jt = wave; jt < N_SEQ / 16; jt += 4) {
      __builtin_prefetch(xn + (size_t)(jt + 4) * 16 * EDIM, 0, 3);
      v16bf bf[8];
#pragma unroll
      for (int kt = 0; kt < EDIM / 32; ++kt)
        bf[kt] = load_b_nk(xn + (size_t)jt * 16 * EDIM + kt * 32, EDIM);
      v8f acc;
#pragma unroll
      for (int r = 0; r < 8; ++r) acc[r] = 0.0f;
#pragma unroll
      for (int kt = 0; kt < EDIM / 32; ++kt) acc = wmma_bf16(af[kt], bf[kt], acc);
      int col = jt * 16 + (lane & 15);
#pragma unroll
      for (int r = 0; r < 8; ++r) {
        int M = r + ((lane < 16) ? 0 : 8);
        S[M * N_SEQ + col] = acc[r];
      }
    }
  }
  __syncthreads();

  // Phase B: row softmax (8 lanes per row, shfl reductions); emit bf16 P strip
  {
    int row = threadIdx.x >> 3, c = threadIdx.x & 7;
    float* Sr = S + row * N_SEQ + c * 128;
    unsigned short* Pr = Pbf + row * N_SEQ + c * 128;
    float mx = -1e30f;
    for (int i = 0; i < 128; ++i) mx = fmaxf(mx, Sr[i]);
#pragma unroll
    for (int o = 1; o < 8; o <<= 1) mx = fmaxf(mx, __shfl_xor(mx, o, 8));
    float sum = 0.0f;
    for (int i = 0; i < 128; ++i) { float e = __expf(Sr[i] - mx); Sr[i] = e; sum += e; }
#pragma unroll
    for (int o = 1; o < 8; o <<= 1) sum += __shfl_xor(sum, o, 8);
    float inv = 1.0f / sum;
    for (int i = 0; i < 128; ++i) Pr[i] = f2bf(Sr[i] * inv);
  }
  __syncthreads();

  // Phase C: g_block = P @ h_b (M=16, K=1024, N=512); one A-load feeds 8 tiles
  const unsigned short* htb = ht + (size_t)b * HID * N_SEQ;
  v8f acc[8];
#pragma unroll
  for (int oi = 0; oi < 8; ++oi)
    for (int r = 0; r < 8; ++r) acc[oi][r] = 0.0f;

  for (int kt = 0; kt < N_SEQ / 32; ++kt) {
    v16bf a = load_a_bf16(Pbf + kt * 32, N_SEQ);      // 2x ds_load_b128
    v16bf bf[8];
#pragma unroll
    for (int oi = 0; oi < 8; ++oi)
      bf[oi] = load_b_nk(htb + (size_t)(wave + oi * 4) * 16 * N_SEQ + kt * 32, N_SEQ);
#pragma unroll
    for (int oi = 0; oi < 8; ++oi)
      acc[oi] = wmma_bf16(a, bf[oi], acc[oi]);
  }
  float localsq = 0.0f;
#pragma unroll
  for (int oi = 0; oi < 8; ++oi) {
    int col = (wave + oi * 4) * 16 + (lane & 15);
#pragma unroll
    for (int r = 0; r < 8; ++r) {
      int M = r + ((lane < 16) ? 0 : 8);
      float v = acc[oi][r];
      localsq += v * v;
      g[(size_t)M * HID + col] = f2bf(v);
    }
  }
  atomicAdd(&sumsq[z], localsq);
}

// ---------------------------------------------------------------------------
// K3: scales = 1/||g||_F
__global__ void k_scales(float* __restrict__ sumsq) {
  if (threadIdx.x < 2) sumsq[2 + threadIdx.x] = rsqrtf(sumsq[threadIdx.x]);
}

// ---------------------------------------------------------------------------
// K4: z = [h, g0*s0, g1*s1] @ W_hg2z + b2   (M=16384, K=1536, N=32)
__global__ void k_final(const unsigned short* __restrict__ h_att,
                        const unsigned short* __restrict__ g0,
                        const unsigned short* __restrict__ g1,
                        const unsigned short* __restrict__ W2t,
                        const float* __restrict__ bias2,
                        const float* __restrict__ scales,
                        float* __restrict__ out) {
  int n = blockIdx.x;                        // rows of the m-tile are batches
  int wave = threadIdx.x >> 5, lane = threadIdx.x & 31;
  int ot = wave;                             // 2 waves, 2 output tiles (TAG=32)
  float s0 = scales[0], s1 = scales[1];

  v8f acc;
#pragma unroll
  for (int r = 0; r < 8; ++r) acc[r] = 0.0f;
  int m = lane & 15;
  for (int kt = 0; kt < 3 * HID / 32; ++kt) {
    v16bf b = load_b_nk(W2t + (size_t)ot * 16 * (3 * HID) + kt * 32, 3 * HID);
    // Build A fragment: rows = batches, K spans [h | g0*s0 | g1*s1]
    U8 ar;
#pragma unroll
    for (int v = 0; v < 8; ++v) {
      int kk = kt * 32 + a_kmap(v, lane);    // even, pair stays in one segment
      int seg = kk >> 9, off = kk & 511;
      const unsigned short* src = (seg == 0) ? h_att : (seg == 1) ? g0 : g1;
      float sc = (seg == 0) ? 1.0f : (seg == 1) ? s0 : s1;
      size_t idx = ((size_t)m * N_SEQ + n) * HID + off;
      float lo = bf2f(src[idx]) * sc;
      float hi = bf2f(src[idx + 1]) * sc;
      ar.u[v] = (unsigned)f2bf(lo) | ((unsigned)f2bf(hi) << 16);
    }
    v16bf a = __builtin_bit_cast(v16bf, ar);
    acc = wmma_bf16(a, b, acc);
  }
  int col = ot * 16 + (lane & 15);
  float bv = bias2[col];
#pragma unroll
  for (int r = 0; r < 8; ++r) {
    int bidx = r + ((lane < 16) ? 0 : 8);
    out[((size_t)n * BATCH + bidx) * TAG + col] = acc[r] + bv;
  }
}

// ---------------------------------------------------------------------------
extern "C" void kernel_launch(void* const* d_in, const int* in_sizes, int n_in,
                              void* d_out, int out_size, void* d_ws, size_t ws_size,
                              hipStream_t stream) {
  const int*   xc = (const int*)d_in[0];
  const int*   x0 = (const int*)d_in[1];
  const int*   x1 = (const int*)d_in[2];
  const float* Ec = (const float*)d_in[3];
  const float* E0 = (const float*)d_in[4];
  const float* E1 = (const float*)d_in[5];
  const float* W1 = (const float*)d_in[6];
  const float* b1 = (const float*)d_in[7];
  const float* W2 = (const float*)d_in[8];
  const float* b2 = (const float*)d_in[9];
  float* out = (float*)d_out;

  char* ws = (char*)d_ws;
  size_t off = 0;
  auto alloc = [&](size_t bytes) {
    size_t r = off;
    off = (off + bytes + 255) & ~(size_t)255;
    return r;
  };
  unsigned short* embed = (unsigned short*)(ws + alloc((size_t)16384 * EMB * 2));
  unsigned short* W1t   = (unsigned short*)(ws + alloc((size_t)HID * EMB * 2));
  unsigned short* W2t   = (unsigned short*)(ws + alloc((size_t)TAG * 3 * HID * 2));
  unsigned short* hbuf  = (unsigned short*)(ws + alloc((size_t)BATCH * N_SEQ * HID * 2));
  unsigned short* htbuf = (unsigned short*)(ws + alloc((size_t)BATCH * HID * N_SEQ * 2));
  unsigned short* xn0   = (unsigned short*)(ws + alloc((size_t)BATCH * N_SEQ * EDIM * 2));
  unsigned short* xn1   = (unsigned short*)(ws + alloc((size_t)BATCH * N_SEQ * EDIM * 2));
  unsigned short* g0    = (unsigned short*)(ws + alloc((size_t)BATCH * N_SEQ * HID * 2));
  unsigned short* g1    = (unsigned short*)(ws + alloc((size_t)BATCH * N_SEQ * HID * 2));
  float*          ssq   = (float*)(ws + alloc(16 * sizeof(float)));

  const int attn_lds = 16 * N_SEQ * 4 + 16 * N_SEQ * 2;   // 96 KB (< 320 KB/WGP)
  (void)hipFuncSetAttribute(reinterpret_cast<const void*>(k_attn),
                            hipFuncAttributeMaxDynamicSharedMemorySize, attn_lds);

  int prep_elems = EMB * HID + 3 * HID * TAG;
  k_prep_w<<<(prep_elems + 255) / 256, 256, 0, stream>>>(W1, W2, W1t, W2t, ssq);
  k_gather<<<16384, 256, 0, stream>>>(xc, x0, x1, Ec, E0, E1, embed, xn0, xn1);
  k_hgemm<<<1024, 128, 0, stream>>>(embed, W1t, b1, hbuf, htbuf);
  k_attn<<<dim3(64, 16, 2), 128, attn_lds, stream>>>(xn0, xn1, htbuf, g0, g1, ssq);
  k_scales<<<1, 32, 0, stream>>>(ssq);
  k_final<<<1024, 64, 0, stream>>>(hbuf, g0, g1, W2t, b2, ssq + 2, out);
}
